// PrimaryNetwork_59064390255265
// MI455X (gfx1250) — matrix operand order repaired and checked by reading, not compiled, not run on hardware
//
#include <hip/hip_runtime.h>
#include <hip/hip_bf16.h>

// ---- problem constants (match reference) ----
static constexpr int Hd   = 512;   // hidden units
static constexpr int OUd  = 64;    // obs units
static constexpr int AUd  = 16;    // act units
static constexpr int Ld   = 2;     // hidden layers
static constexpr int Bd   = 512;   // obs batch
static constexpr int Gd   = 128;   // gate samples
static constexpr int GDIM = 3 * Hd + Ld * Hd + OUd + 2;   // 2626

static constexpr int ROWS = 64;    // rows (g samples) per block
static constexpr int XS0  = 1032;  // padded stride for K=1024 buffer
static constexpr int XS1  = 520;   // padded stride for K=512 buffer
static constexpr int FRAG = 512;   // f16 elements per B fragment (32 lanes x 16)

typedef __attribute__((ext_vector_type(16))) _Float16 v16h;
typedef __attribute__((ext_vector_type(8)))  _Float16 v8h;
typedef __attribute__((ext_vector_type(8)))  float    v8f;

// ---------------------------------------------------------------------------
// Prep kernels
// ---------------------------------------------------------------------------

__global__ void prep_obsn_kernel(const float* __restrict__ obs,
                                 const float* __restrict__ omin,
                                 const float* __restrict__ omax,
                                 float* __restrict__ obs_n) {
    int idx = blockIdx.x * blockDim.x + threadIdx.x;   // B*OU threads
    int k = idx & (OUd - 1);
    float t = (obs[idx] - omin[k]) / (omax[k] - omin[k]);
    obs_n[idx] = 2.0f * t - 1.0f;                      // OBS_SCALE == 1
}

__global__ void prep_gexp_kernel(const float* __restrict__ g,
                                 float* __restrict__ gexp, int n) {
    int idx = blockIdx.x * blockDim.x + threadIdx.x;
    if (idx < n) gexp[idx] = __expf(g[idx]);
}

// row-wise weight normalization -> f16 copy.  One wave (32 lanes) per row.
__global__ void wn_rows_kernel(const float* __restrict__ src,
                               _Float16* __restrict__ dst,
                               int rows, int cols) {
    int wv = threadIdx.x >> 5, lane = threadIdx.x & 31;
    int row = blockIdx.x * 8 + wv;
    if (row >= rows) return;
    const float* s = src + (size_t)row * cols;
    float ss = 0.f;
    for (int k = lane; k < cols; k += 32) { float v = s[k]; ss += v * v; }
    #pragma unroll
    for (int off = 16; off >= 1; off >>= 1) ss += __shfl_xor(ss, off, 32);
    float inv = rsqrtf(ss);
    _Float16* d = dst + (size_t)row * cols;
    for (int k = lane; k < cols; k += 32) d[k] = (_Float16)(s[k] * inv);
}

// Swizzle row-major W[rows][K] (f16) into WMMA B-fragment order:
// fragment f = (n/16)*(K/32) + (k/32) holds 32 lanes x 16 f16 (1KB contiguous);
// lane = (n%16) + ((k/16)%2)*16 ; elem = k%16.
// Hot-loop loads then become fully coalesced b128 pairs.
__global__ void swz_b_kernel(const _Float16* __restrict__ src,
                             _Float16* __restrict__ dst,
                             int rows, int K) {
    int idx = blockIdx.x * blockDim.x + threadIdx.x;
    if (idx >= rows * K) return;
    int n = idx / K, k = idx - n * K;
    int f    = (n >> 4) * (K >> 5) + (k >> 5);
    int lane = (n & 15) + (((k >> 4) & 1) << 4);
    int i    = k & 15;
    dst[(size_t)f * FRAG + lane * 16 + i] = src[idx];
}

// out[b][h] = sum_k x[b][k] * w16[h][k]   (tiny: K = 64 or 16)
__global__ void pre_gemm_kernel(const float* __restrict__ x,
                                const _Float16* __restrict__ w16,
                                float* __restrict__ out, int K) {
    int idx = blockIdx.x * blockDim.x + threadIdx.x;   // B*H threads
    int b = idx >> 9;            // /Hd
    int h = idx & (Hd - 1);
    const float*    xr = x   + (size_t)b * K;
    const _Float16* wr = w16 + (size_t)h * K;
    float s = 0.f;
    #pragma unroll 8
    for (int k = 0; k < K; ++k) s += xr[k] * (float)wr[k];
    out[idx] = s;
}

// ---------------------------------------------------------------------------
// Fused MLP kernel
// ---------------------------------------------------------------------------

__device__ __forceinline__ v16h cat8(v8h lo, v8h hi) {
    return __builtin_shufflevector(lo, hi, 0,1,2,3,4,5,6,7,8,9,10,11,12,13,14,15);
}

__device__ __forceinline__ v8f wmma16(v16h a, v16h b, v8f c) {
    return __builtin_amdgcn_wmma_f32_16x16x32_f16(false, a, false, b,
                                                  (short)0, c, false, false);
}

// A fragment in contiguous k order (k = 16*(lane>>4) + elem), same as swz_b: lane = row (lane&15),
// K halves selected by lane>>4 (handled via kbase by caller).
__device__ __forceinline__ v16h loadA(const _Float16* __restrict__ X, int xs,
                                      int row, int kbase) {
    const _Float16* p = X + (size_t)row * xs + kbase;
    return cat8(*(const v8h*)(p), *(const v8h*)(p + 8));
}

// B fragment from swizzled weights: contiguous 1KB per fragment, lane-major.
__device__ __forceinline__ v16h loadB(const _Float16* __restrict__ wbase, int kb) {
    const _Float16* p = wbase + (size_t)kb * FRAG;
    return cat8(*(const v8h*)(p), *(const v8h*)(p + 8));
}

// One layer over the 64-row tile:
//   Y[64][512] = act( (X[64][K] @ W[512][K]^T) * gate + bias )
// Col-block-major wave assignment: wave wv owns col blocks {wv, wv+8, wv+16, wv+24};
// for each col block it holds 4 row-block accumulators and loads each B fragment
// exactly once -> the whole block reads W exactly once, fully coalesced.
template <int K, bool RELU>
__device__ __forceinline__ void do_layer(const _Float16* __restrict__ X, int xs,
                                         _Float16* __restrict__ Y, int ys,
                                         const _Float16* __restrict__ Wsw,
                                         const float* __restrict__ gexp, int gate_off,
                                         const float* __restrict__ bias,
                                         int g0, int lane, int wv) {
    const int asel = lane >> 4;          // 0|1 : K-half select for A layout
    const int nloc = lane & 15;

    for (int c = 0; c < 4; ++c) {
        const int nb = wv + c * 8;                       // 0..31
        const int n  = nb * 16 + nloc;
        const _Float16* wbase = Wsw + ((size_t)nb * (K / 32)) * FRAG + lane * 16;
        v8f acc0 = {}, acc1 = {}, acc2 = {}, acc3 = {};
        #pragma unroll 2
        for (int kb = 0; kb < K / 32; ++kb) {
            v16h bb = loadB(wbase, kb);
            const int ka = kb * 32 + asel * 16;
            acc0 = wmma16(loadA(X, xs, 0 * 16 + nloc, ka), bb, acc0);
            acc1 = wmma16(loadA(X, xs, 1 * 16 + nloc, ka), bb, acc1);
            acc2 = wmma16(loadA(X, xs, 2 * 16 + nloc, ka), bb, acc2);
            acc3 = wmma16(loadA(X, xs, 3 * 16 + nloc, ka), bb, acc3);
        }
        const float bn = bias[n];
        #pragma unroll
        for (int rb = 0; rb < 4; ++rb) {
            v8f acc = (rb == 0) ? acc0 : (rb == 1) ? acc1 : (rb == 2) ? acc2 : acc3;
            #pragma unroll
            for (int r = 0; r < 8; ++r) {
                int m  = rb * 16 + r + asel * 8;   // C layout: lanes16-31 -> M=8+r
                int gg = g0 + m;
                float v = acc[r] * gexp[(size_t)gg * GDIM + gate_off + n] + bn;
                if (RELU) v = v > 0.f ? v : 0.01f * v;
                Y[(size_t)m * ys + n] = (_Float16)v;
            }
        }
    }
}

__global__ __launch_bounds__(256)
void fused_mlp_kernel(const float* __restrict__ gexp,
                      const float* __restrict__ o_pre,
                      const float* __restrict__ a_pre,
                      const float* __restrict__ obs_n,
                      const _Float16* __restrict__ w_tgt_sw,
                      const _Float16* __restrict__ w_hid_sw,
                      const _Float16* __restrict__ w_st_sw,
                      const _Float16* __restrict__ w_rw,
                      const _Float16* __restrict__ w_dn,
                      const float* __restrict__ obs_b,
                      const float* __restrict__ act_b,
                      const float* __restrict__ tgt_b,
                      const float* __restrict__ hidden_b,
                      const float* __restrict__ state_b,
                      const float* __restrict__ reward_b,
                      const float* __restrict__ done_b,
                      const float* __restrict__ omin,
                      const float* __restrict__ omax,
                      float* __restrict__ out_states,
                      float* __restrict__ out_rew,
                      float* __restrict__ out_done) {
    // 64-row tile: one b, 64 consecutive g
    const int b  = blockIdx.x >> 1;
    const int g0 = (blockIdx.x & 1) * ROWS;
    const int tid  = threadIdx.x;
    const int lane = tid & 31;
    const int wv   = tid >> 5;

    __shared__ __align__(16) _Float16 smemA[ROWS * XS0];   // K=1024 buf (padded)
    __shared__ __align__(16) _Float16 smemB[ROWS * XS1];   // K=512  buf (padded)

    // ---- layer 0: y0 = lrelu([o_pre*g_obs+obs_b , a_pre*g_act+act_b]) ----
    for (int idx = tid; idx < ROWS * 1024; idx += 256) {
        int r = idx >> 10, k = idx & 1023;
        int gg = g0 + r;
        float v;
        if (k < Hd) {
            v = o_pre[(size_t)b * Hd + k] * gexp[(size_t)gg * GDIM + k] + obs_b[k];
        } else {
            int kk = k - Hd;
            v = a_pre[(size_t)b * Hd + kk] * gexp[(size_t)gg * GDIM + Hd + kk] + act_b[kk];
        }
        v = v > 0.f ? v : 0.01f * v;
        smemA[r * XS0 + k] = (_Float16)v;
    }
    __syncthreads();

    // ---- tgt layer: [64x1024] -> [64x512] ----
    do_layer<1024, true>(smemA, XS0, smemB, XS1, w_tgt_sw, gexp, 2 * Hd, tgt_b,
                         g0, lane, wv);
    __syncthreads();
    // ---- hidden 0 ----
    do_layer<512, true>(smemB, XS1, smemA, XS1, w_hid_sw, gexp, 3 * Hd, hidden_b,
                        g0, lane, wv);
    __syncthreads();
    // ---- hidden 1 ----
    do_layer<512, true>(smemA, XS1, smemB, XS1, w_hid_sw + (size_t)Hd * Hd, gexp,
                        3 * Hd + Hd, hidden_b + Hd, g0, lane, wv);
    __syncthreads();

    // ---- state head: [64x512] -> [64x64], 16 tiles -> 2 per wave ----
    {
        const int cb    = wv & 3;            // col block 0..3
        const int rbase = (wv >> 2) * 2;     // row blocks {0,1} or {2,3}
        const int asel  = lane >> 4;
        const int nloc  = lane & 15;
        const int n     = cb * 16 + nloc;
        const _Float16* wbase = w_st_sw + ((size_t)cb * (Hd / 32)) * FRAG + lane * 16;
        const float sc = omax[n] - omin[n];
        #pragma unroll
        for (int t = 0; t < 2; ++t) {
            const int rb = rbase + t;
            v8f acc = {};
            #pragma unroll 4
            for (int kb = 0; kb < Hd / 32; ++kb) {
                acc = wmma16(loadA(smemB, XS1, rb * 16 + nloc, kb * 32 + asel * 16),
                             loadB(wbase, kb), acc);
            }
            #pragma unroll
            for (int r = 0; r < 8; ++r) {
                int m  = rb * 16 + r + asel * 8;
                int gg = g0 + m;
                int row = b * Gd + gg;
                float v = acc[r] * gexp[(size_t)gg * GDIM + (3 * Hd + Ld * Hd) + n]
                          + state_b[n] + obs_n[(size_t)b * OUd + n];
                v = (v + 1.0f) * 0.5f;          // OBS_SCALE == 1
                v = v * sc + omin[n];
                out_states[(size_t)row * OUd + n] = v;
            }
        }
    }

    // ---- reward / done heads (N=1): wave-level dot products, 8 rows/wave ----
    {
        const int i0 = 3 * Hd + Ld * Hd;     // 2560
        for (int mr = 0; mr < 8; ++mr) {
            int m = wv * 8 + mr;
            float sr = 0.f, sd = 0.f;
            int k0 = lane * 16;
            #pragma unroll
            for (int i = 0; i < 16; ++i) {
                float y = (float)smemB[(size_t)m * XS1 + k0 + i];
                sr += y * (float)w_rw[k0 + i];
                sd += y * (float)w_dn[k0 + i];
            }
            #pragma unroll
            for (int off = 16; off >= 1; off >>= 1) {
                sr += __shfl_xor(sr, off, 32);
                sd += __shfl_xor(sd, off, 32);
            }
            if (lane == 0) {
                int gg = g0 + m;
                int row = b * Gd + gg;
                float rv = sr * gexp[(size_t)gg * GDIM + i0 + OUd]     + reward_b[0];
                float dv = sd * gexp[(size_t)gg * GDIM + i0 + OUd + 1] + done_b[0];
                dv = 1.0f / (1.0f + __expf(-dv));
                out_rew[row]  = rv;
                out_done[row] = dv;
            }
        }
    }
}

// ---------------------------------------------------------------------------
// Host launch
// ---------------------------------------------------------------------------

extern "C" void kernel_launch(void* const* d_in, const int* in_sizes, int n_in,
                              void* d_out, int out_size, void* d_ws, size_t ws_size,
                              hipStream_t stream) {
    // inputs (setup_inputs order)
    const float* obs      = (const float*)d_in[0];
    const float* action   = (const float*)d_in[1];
    const float* g        = (const float*)d_in[2];
    const float* obs_w    = (const float*)d_in[3];
    const float* obs_b    = (const float*)d_in[4];
    const float* act_w    = (const float*)d_in[5];
    const float* act_b    = (const float*)d_in[6];
    const float* tgt_w    = (const float*)d_in[7];
    const float* tgt_b    = (const float*)d_in[8];
    const float* hidden_w = (const float*)d_in[9];
    const float* hidden_b = (const float*)d_in[10];
    const float* state_w  = (const float*)d_in[11];
    const float* state_b  = (const float*)d_in[12];
    const float* reward_w = (const float*)d_in[13];
    const float* reward_b = (const float*)d_in[14];
    const float* done_w   = (const float*)d_in[15];
    const float* done_b   = (const float*)d_in[16];
    const float* obs_min  = (const float*)d_in[17];
    const float* obs_max  = (const float*)d_in[18];

    // workspace layout (bytes)
    char* ws = (char*)d_ws;
    size_t off = 0;
    auto alloc = [&](size_t bytes) { char* p = ws + off; off += (bytes + 255) & ~size_t(255); return p; };
    float*    obs_n    = (float*)   alloc((size_t)Bd * OUd * 4);
    float*    gexp     = (float*)   alloc((size_t)Gd * GDIM * 4);
    float*    o_pre    = (float*)   alloc((size_t)Bd * Hd * 4);
    float*    a_pre    = (float*)   alloc((size_t)Bd * Hd * 4);
    _Float16* w_obs    = (_Float16*)alloc((size_t)Hd * OUd * 2);
    _Float16* w_act    = (_Float16*)alloc((size_t)Hd * AUd * 2);
    _Float16* w_tgt    = (_Float16*)alloc((size_t)Hd * 2 * Hd * 2);
    _Float16* w_hid    = (_Float16*)alloc((size_t)Ld * Hd * Hd * 2);
    _Float16* w_st     = (_Float16*)alloc((size_t)OUd * Hd * 2);
    _Float16* w_rw     = (_Float16*)alloc((size_t)Hd * 2);
    _Float16* w_dn     = (_Float16*)alloc((size_t)Hd * 2);
    _Float16* w_tgt_sw = (_Float16*)alloc((size_t)Hd * 2 * Hd * 2);
    _Float16* w_hid_sw = (_Float16*)alloc((size_t)Ld * Hd * Hd * 2);
    _Float16* w_st_sw  = (_Float16*)alloc((size_t)OUd * Hd * 2);
    (void)ws_size; (void)n_in; (void)in_sizes; (void)out_size;

    // outputs: [next_states | rewards | dones] flat
    float* out_states = (float*)d_out;
    float* out_rew    = out_states + (size_t)Bd * Gd * OUd;
    float* out_done   = out_rew    + (size_t)Bd * Gd;

    // 1) normalize observations
    prep_obsn_kernel<<<(Bd * OUd) / 256, 256, 0, stream>>>(obs, obs_min, obs_max, obs_n);
    // 2) exp(gates)
    {
        int n = Gd * GDIM;
        prep_gexp_kernel<<<(n + 255) / 256, 256, 0, stream>>>(g, gexp, n);
    }
    // 3) weight-normalize + f16 convert all matrices
    wn_rows_kernel<<<(Hd + 7) / 8, 256, 0, stream>>>(obs_w, w_obs, Hd, OUd);
    wn_rows_kernel<<<(Hd + 7) / 8, 256, 0, stream>>>(act_w, w_act, Hd, AUd);
    wn_rows_kernel<<<(Hd + 7) / 8, 256, 0, stream>>>(tgt_w, w_tgt, Hd, 2 * Hd);
    wn_rows_kernel<<<(Ld * Hd + 7) / 8, 256, 0, stream>>>(hidden_w, w_hid, Ld * Hd, Hd);
    wn_rows_kernel<<<(OUd + 7) / 8, 256, 0, stream>>>(state_w, w_st, OUd, Hd);
    wn_rows_kernel<<<1, 256, 0, stream>>>(reward_w, w_rw, 1, Hd);
    wn_rows_kernel<<<1, 256, 0, stream>>>(done_w, w_dn, 1, Hd);
    // 4) swizzle WMMA-consumed weights into coalesced B-fragment order
    swz_b_kernel<<<(Hd * 2 * Hd + 255) / 256, 256, 0, stream>>>(w_tgt, w_tgt_sw, Hd, 2 * Hd);
    swz_b_kernel<<<(Hd * Hd + 255) / 256, 256, 0, stream>>>(w_hid, w_hid_sw, Hd, Hd);
    swz_b_kernel<<<(Hd * Hd + 255) / 256, 256, 0, stream>>>(w_hid + (size_t)Hd * Hd,
                                                            w_hid_sw + (size_t)Hd * Hd, Hd, Hd);
    swz_b_kernel<<<(OUd * Hd + 255) / 256, 256, 0, stream>>>(w_st, w_st_sw, OUd, Hd);
    // 5) o_pre = obs_n @ wn(obs_w)^T ; a_pre = action @ wn(act_w)^T
    pre_gemm_kernel<<<(Bd * Hd) / 256, 256, 0, stream>>>(obs_n, w_obs, o_pre, OUd);
    pre_gemm_kernel<<<(Bd * Hd) / 256, 256, 0, stream>>>(action, w_act, a_pre, AUd);
    // 6) fused WMMA MLP over all B*G rows (64 g's per block)
    fused_mlp_kernel<<<Bd * (Gd / ROWS), 256, 0, stream>>>(
        gexp, o_pre, a_pre, obs_n,
        w_tgt_sw, w_hid_sw, w_st_sw, w_rw, w_dn,
        obs_b, act_b, tgt_b, hidden_b, state_b, reward_b, done_b,
        obs_min, obs_max,
        out_states, out_rew, out_done);
}
